// TransformerBlock_81295140978808
// MI455X (gfx1250) — matrix operand order repaired and checked by reading, not compiled, not run on hardware
//
#include <hip/hip_runtime.h>

// ---------------- CDNA5 / gfx1250 types ----------------
typedef __bf16 bf16;
typedef __attribute__((ext_vector_type(16))) __bf16 v16bf;
typedef __attribute__((ext_vector_type(8)))  __bf16 v8bf;
typedef __attribute__((ext_vector_type(8)))  float  v8f;

union Frag  { v16bf v; uint4 q[2]; };   // 32B WMMA bf16 A/B fragment
union Pack8 { v8bf  v; uint4 q;    };   // 16B packed bf16 store

#define D_MODEL 1024
#define SEQ     2048
#define NHEAD   16
#define HDIM    64
#define MTOT    4096      // B*S
#define DFF     4096

__device__ __forceinline__ v8f v8f_zero() {
  v8f z;
  #pragma unroll
  for (int i = 0; i < 8; ++i) z[i] = 0.0f;
  return z;
}

// ---------------- DPP16 cross-lane helpers (pure VALU, no LDS) -------------
// CTRL: 0xB1 = quad_perm(1,0,3,2) -> xor1 ; 0x4E = quad_perm(2,3,0,1) -> xor2
//       0x141 = ROW_HALF_MIRROR (xor4 once 4-groups uniform)
//       0x140 = ROW_MIRROR      (xor8 once 8-groups uniform)
template <int CTRL>
__device__ __forceinline__ float dpp_f32(float v) {
  int x = __builtin_bit_cast(int, v);
  int y = __builtin_amdgcn_update_dpp(x, x, CTRL, 0xF, 0xF, true);
  return __builtin_bit_cast(float, y);
}

// reduce across the 16-lane half (all 16 lanes end with the result)
__device__ __forceinline__ float redmax16(float v) {
  v = fmaxf(v, dpp_f32<0xB1>(v));
  v = fmaxf(v, dpp_f32<0x4E>(v));
  v = fmaxf(v, dpp_f32<0x141>(v));
  v = fmaxf(v, dpp_f32<0x140>(v));
  return v;
}
__device__ __forceinline__ float redsum16(float v) {
  v += dpp_f32<0xB1>(v);
  v += dpp_f32<0x4E>(v);
  v += dpp_f32<0x141>(v);
  v += dpp_f32<0x140>(v);
  return v;
}

// ---------------------------------------------------------------------------
// Weight prep: W[K][N] f32 row-major  ->  Wt[N][K] bf16 row-major
// ---------------------------------------------------------------------------
__global__ __launch_bounds__(256) void transpose_f32_to_bf16(
    const float* __restrict__ W, bf16* __restrict__ Wt, int K, int N)
{
  __shared__ float tile[32][33];
  int n0 = blockIdx.x * 32, k0 = blockIdx.y * 32;
  int tx = threadIdx.x, ty = threadIdx.y;   // 32 x 8
  #pragma unroll
  for (int r = 0; r < 32; r += 8)
    tile[ty + r][tx] = W[(size_t)(k0 + ty + r) * N + n0 + tx];
  __syncthreads();
  #pragma unroll
  for (int r = 0; r < 32; r += 8)
    Wt[(size_t)(n0 + ty + r) * K + k0 + tx] = (bf16)tile[tx][ty + r];
}

// ---------------------------------------------------------------------------
// LayerNorm(x; g, be) -> bf16 out.  One row (D=1024) per block, 256 threads.
// ---------------------------------------------------------------------------
__global__ __launch_bounds__(256) void ln_to_bf16_kernel(
    const float* __restrict__ x, const float* __restrict__ g,
    const float* __restrict__ be, bf16* __restrict__ out)
{
  int row = blockIdx.x;
  const float* xr = x + (size_t)row * D_MODEL;
  float v[4], s = 0.f, s2 = 0.f;
  #pragma unroll
  for (int j = 0; j < 4; ++j) {
    v[j] = xr[threadIdx.x + 256 * j];
    s += v[j]; s2 += v[j] * v[j];
  }
  __shared__ float sh1[256], sh2[256];
  sh1[threadIdx.x] = s; sh2[threadIdx.x] = s2;
  __syncthreads();
  for (int off = 128; off > 0; off >>= 1) {
    if (threadIdx.x < off) {
      sh1[threadIdx.x] += sh1[threadIdx.x + off];
      sh2[threadIdx.x] += sh2[threadIdx.x + off];
    }
    __syncthreads();
  }
  float mu   = sh1[0] * (1.f / D_MODEL);
  float var  = sh2[0] * (1.f / D_MODEL) - mu * mu;
  float rstd = rsqrtf(var + 1e-5f);
  #pragma unroll
  for (int j = 0; j < 4; ++j) {
    int c = threadIdx.x + 256 * j;
    out[(size_t)row * D_MODEL + c] = (bf16)((v[j] - mu) * rstd * g[c] + be[c]);
  }
}

// ---------------------------------------------------------------------------
// x_res = attn + x  (f32 + bf16 copies), x_norm2 = LN(x_res; g2, be2) (f32)
// ---------------------------------------------------------------------------
__global__ __launch_bounds__(256) void resid_ln2_kernel(
    const float* __restrict__ attn, const float* __restrict__ x,
    const float* __restrict__ g, const float* __restrict__ be,
    float* __restrict__ xres, bf16* __restrict__ xres_bf,
    float* __restrict__ xnorm2)
{
  int row = blockIdx.x;
  size_t base = (size_t)row * D_MODEL;
  float v[4], s = 0.f, s2 = 0.f;
  #pragma unroll
  for (int j = 0; j < 4; ++j) {
    int c = threadIdx.x + 256 * j;
    v[j] = attn[base + c] + x[base + c];
    xres[base + c]    = v[j];
    xres_bf[base + c] = (bf16)v[j];
    s += v[j]; s2 += v[j] * v[j];
  }
  __shared__ float sh1[256], sh2[256];
  sh1[threadIdx.x] = s; sh2[threadIdx.x] = s2;
  __syncthreads();
  for (int off = 128; off > 0; off >>= 1) {
    if (threadIdx.x < off) {
      sh1[threadIdx.x] += sh1[threadIdx.x + off];
      sh2[threadIdx.x] += sh2[threadIdx.x + off];
    }
    __syncthreads();
  }
  float mu   = sh1[0] * (1.f / D_MODEL);
  float var  = sh2[0] * (1.f / D_MODEL) - mu * mu;
  float rstd = rsqrtf(var + 1e-5f);
  #pragma unroll
  for (int j = 0; j < 4; ++j) {
    int c = threadIdx.x + 256 * j;
    xnorm2[base + c] = (v[j] - mu) * rstd * g[c] + be[c];
  }
}

// ---------------------------------------------------------------------------
// WMMA bf16 GEMM:  C[M,N] = A[M,K] @ B[K,N],  B given pre-transposed Bt[N,K].
// Block = 128 threads (4 waves); block tile 128x128; wave tile 64x64
// (16 accumulators, 16 v_wmma_f32_16x16x32_bf16 per K-step of 32).
// MODE 0: store bf16 [M,N]
// MODE 1: store bf16 transposed per-head (V): Vt[(b*H+h)*64+hd][s]
// MODE 2: store bf16 relu(acc + bias[n])
// MODE 3: store f32  acc + bias[n] + addm[m,n]
// ---------------------------------------------------------------------------
template <int MODE>
__global__ __launch_bounds__(128) void gemm_bf16_kernel(
    const bf16* __restrict__ A, const bf16* __restrict__ Bt,
    int M, int N, int K,
    const float* __restrict__ bias, const float* __restrict__ addm,
    bf16* __restrict__ outb, float* __restrict__ outf)
{
  const int lane = threadIdx.x & 31;
  const int wv   = threadIdx.x >> 5;
  const int hf   = lane >> 4;     // lane half (0/1)
  const int l16  = lane & 15;
  const int m0 = blockIdx.y * 128 + (wv >> 1) * 64;
  const int n0 = blockIdx.x * 128 + (wv & 1) * 64;

  v8f acc[4][4];
  #pragma unroll
  for (int mt = 0; mt < 4; ++mt)
    #pragma unroll
    for (int nt = 0; nt < 4; ++nt) acc[mt][nt] = v8f_zero();

  for (int k = 0; k < K; k += 32) {
    Frag a[4], b[4];
    #pragma unroll
    for (int mt = 0; mt < 4; ++mt) {
      const bf16* ap = A + (size_t)(m0 + mt * 16 + l16) * K + k + 8 * hf;
      a[mt].q[0] = *(const uint4*)(ap);
      a[mt].q[1] = *(const uint4*)(ap + 16);
    }
    #pragma unroll
    for (int nt = 0; nt < 4; ++nt) {
      const bf16* bp = Bt + (size_t)(n0 + nt * 16 + l16) * K + k + 8 * hf;
      b[nt].q[0] = *(const uint4*)(bp);
      b[nt].q[1] = *(const uint4*)(bp + 16);
    }
    #pragma unroll
    for (int mt = 0; mt < 4; ++mt)
      #pragma unroll
      for (int nt = 0; nt < 4; ++nt)
        acc[mt][nt] = __builtin_amdgcn_wmma_f32_16x16x32_bf16(
            false, a[mt].v, false, b[nt].v, (short)0, acc[mt][nt], false, false);
  }

  // epilogue: C layout = lane(n = l16, half picks M 0-7 / 8-15), vgpr i -> m
  #pragma unroll
  for (int mt = 0; mt < 4; ++mt) {
    #pragma unroll
    for (int nt = 0; nt < 4; ++nt) {
      if (MODE == 1) {
        // transposed per-head store for V: contiguous 8 bf16 along s
        Pack8 pk;
        #pragma unroll
        for (int i = 0; i < 8; ++i) pk.v[i] = (bf16)acc[mt][nt][i];
        int mg = m0 + mt * 16 + 8 * hf;       // token row (s within batch)
        int n  = n0 + nt * 16 + l16;          // d_model column
        size_t idx = ((size_t)(mg >> 11) * D_MODEL + n) * SEQ + (mg & (SEQ - 1));
        *(uint4*)(outb + idx) = pk.q;
      } else {
        int n = n0 + nt * 16 + l16;
        #pragma unroll
        for (int i = 0; i < 8; ++i) {
          int m = m0 + mt * 16 + i + 8 * hf;
          float val = acc[mt][nt][i];
          if (MODE == 0) {
            outb[(size_t)m * N + n] = (bf16)val;
          } else if (MODE == 2) {
            val += bias[n];
            outb[(size_t)m * N + n] = (bf16)fmaxf(val, 0.0f);
          } else { // MODE 3
            outf[(size_t)m * N + n] = val + bias[n] + addm[(size_t)m * N + n];
          }
        }
      }
    }
  }
}

// ---------------------------------------------------------------------------
// Flash attention (per wave: 16 query rows x Hd=64, online softmax).
// Qb, Kb: bf16 [B*S, D] (per-head columns). Vt: bf16 [B*H*Hd, S] transposed.
// Row max/sum reductions are DPP16-only (no LDS, co-executes with WMMA).
// ---------------------------------------------------------------------------
__global__ __launch_bounds__(128) void attention_kernel(
    const bf16* __restrict__ Qb, const bf16* __restrict__ Kb,
    const bf16* __restrict__ Vt, float* __restrict__ attnOut)
{
  __shared__ __align__(16) bf16 ldsP[4][16][32];   // per-wave P staging
  const int lane = threadIdx.x & 31;
  const int wv   = threadIdx.x >> 5;
  const int gid  = blockIdx.x * 4 + wv;            // q-tile id
  const int bat  = gid >> 11;                      // / (H * S/16)
  const int rem  = gid & 2047;
  const int hed  = rem >> 7;
  const int q0   = (rem & 127) << 4;
  const int hf   = lane >> 4;
  const int l16  = lane & 15;

  // Q tile as two A-fragments (d 0..31, 32..63)
  Frag aq0, aq1;
  {
    const bf16* qp = Qb + (size_t)(bat * SEQ + q0 + l16) * D_MODEL + hed * HDIM + 8 * hf;
    aq0.q[0] = *(const uint4*)(qp);
    aq0.q[1] = *(const uint4*)(qp + 16);
    aq1.q[0] = *(const uint4*)(qp + 32);
    aq1.q[1] = *(const uint4*)(qp + 48);
  }

  v8f o[4];
  #pragma unroll
  for (int nt = 0; nt < 4; ++nt) o[nt] = v8f_zero();
  float mrun[8], lrun[8];
  #pragma unroll
  for (int i = 0; i < 8; ++i) { mrun[i] = -1e30f; lrun[i] = 0.0f; }

  for (int kv0 = 0; kv0 < SEQ; kv0 += 32) {
    // scores: two 16x16 tiles (kv columns kv0.., kv0+16..)
    v8f s0, s1;
    {
      const bf16* kp = Kb + (size_t)(bat * SEQ + kv0 + l16) * D_MODEL + hed * HDIM + 8 * hf;
      Frag b0, b1;
      b0.q[0] = *(const uint4*)(kp);      b0.q[1] = *(const uint4*)(kp + 16);
      b1.q[0] = *(const uint4*)(kp + 32); b1.q[1] = *(const uint4*)(kp + 48);
      v8f z = v8f_zero();
      z = __builtin_amdgcn_wmma_f32_16x16x32_bf16(false, aq0.v, false, b0.v, (short)0, z, false, false);
      z = __builtin_amdgcn_wmma_f32_16x16x32_bf16(false, aq1.v, false, b1.v, (short)0, z, false, false);
      s0 = z;
    }
    {
      const bf16* kp = Kb + (size_t)(bat * SEQ + kv0 + 16 + l16) * D_MODEL + hed * HDIM + 8 * hf;
      Frag b0, b1;
      b0.q[0] = *(const uint4*)(kp);      b0.q[1] = *(const uint4*)(kp + 16);
      b1.q[0] = *(const uint4*)(kp + 32); b1.q[1] = *(const uint4*)(kp + 48);
      v8f z = v8f_zero();
      z = __builtin_amdgcn_wmma_f32_16x16x32_bf16(false, aq0.v, false, b0.v, (short)0, z, false, false);
      z = __builtin_amdgcn_wmma_f32_16x16x32_bf16(false, aq1.v, false, b1.v, (short)0, z, false, false);
      s1 = z;
    }

    // online softmax: rows live in lane-halves; DPP16 reductions (VALU only)
    float alpha[8];
    #pragma unroll
    for (int i = 0; i < 8; ++i) {
      float v0 = s0[i] * 0.125f;           // 1/sqrt(64)
      float v1 = s1[i] * 0.125f;
      float tm = redmax16(fmaxf(v0, v1));
      float nm = fmaxf(mrun[i], tm);
      float al = __expf(mrun[i] - nm);
      float p0 = __expf(v0 - nm);
      float p1 = __expf(v1 - nm);
      float rs = redsum16(p0 + p1);
      lrun[i] = lrun[i] * al + rs;
      mrun[i] = nm;
      alpha[i] = al;
      ldsP[wv][i + 8 * hf][l16]      = (bf16)p0;
      ldsP[wv][i + 8 * hf][16 + l16] = (bf16)p1;
    }
    asm volatile("" ::: "memory");   // keep LDS store->load order (HW is in-order)

    #pragma unroll
    for (int nt = 0; nt < 4; ++nt)
      #pragma unroll
      for (int i = 0; i < 8; ++i) o[nt][i] *= alpha[i];

    // P back as A-fragment (16 x 32)
    Frag ap;
    ap.q[0] = *(const uint4*)&ldsP[wv][l16][8 * hf];
    ap.q[1] = *(const uint4*)&ldsP[wv][l16][16 + 8 * hf];
    asm volatile("" ::: "memory");

    // o += P @ V  (B fragments contiguous from transposed V)
    #pragma unroll
    for (int nt = 0; nt < 4; ++nt) {
      const bf16* vp = Vt + (size_t)((bat * NHEAD + hed) * HDIM + nt * 16 + l16) * SEQ
                          + kv0 + 8 * hf;
      Frag bv;
      bv.q[0] = *(const uint4*)(vp);
      bv.q[1] = *(const uint4*)(vp + 16);
      o[nt] = __builtin_amdgcn_wmma_f32_16x16x32_bf16(
          false, ap.v, false, bv.v, (short)0, o[nt], false, false);
    }
  }

  #pragma unroll
  for (int i = 0; i < 8; ++i) {
    float inv = 1.0f / lrun[i];
    size_t rowbase = (size_t)(bat * SEQ + q0 + i + 8 * hf) * D_MODEL + hed * HDIM;
    #pragma unroll
    for (int nt = 0; nt < 4; ++nt)
      attnOut[rowbase + nt * 16 + l16] = o[nt][i] * inv;
  }
}

// ---------------------------------------------------------------------------
// Host: orchestrate the block.
// Inputs: x, Wq, Wk, Wv, W1, b1, W2, b2, g1, be1, g2, be2
// ---------------------------------------------------------------------------
extern "C" void kernel_launch(void* const* d_in, const int* in_sizes, int n_in,
                              void* d_out, int out_size, void* d_ws, size_t ws_size,
                              hipStream_t stream) {
  (void)in_sizes; (void)n_in; (void)out_size; (void)ws_size;
  const float* x   = (const float*)d_in[0];
  const float* Wq  = (const float*)d_in[1];
  const float* Wk  = (const float*)d_in[2];
  const float* Wv  = (const float*)d_in[3];
  const float* W1  = (const float*)d_in[4];
  const float* b1  = (const float*)d_in[5];
  const float* W2  = (const float*)d_in[6];
  const float* b2  = (const float*)d_in[7];
  const float* g1  = (const float*)d_in[8];
  const float* be1 = (const float*)d_in[9];
  const float* g2  = (const float*)d_in[10];
  const float* be2 = (const float*)d_in[11];

  char* wp = (char*)d_ws;
  auto take = [&](size_t bytes) {
    char* r = wp;
    wp += (bytes + 255) & ~(size_t)255;
    return r;
  };

  bf16*  WqT    = (bf16*)take((size_t)D_MODEL * D_MODEL * 2);
  bf16*  WkT    = (bf16*)take((size_t)D_MODEL * D_MODEL * 2);
  bf16*  WvT    = (bf16*)take((size_t)D_MODEL * D_MODEL * 2);
  bf16*  W1T    = (bf16*)take((size_t)DFF * D_MODEL * 2);
  bf16*  W2T    = (bf16*)take((size_t)D_MODEL * DFF * 2);
  bf16*  xn     = (bf16*)take((size_t)MTOT * D_MODEL * 2);
  bf16*  Qb     = (bf16*)take((size_t)MTOT * D_MODEL * 2);
  bf16*  Kb     = (bf16*)take((size_t)MTOT * D_MODEL * 2);
  bf16*  Vt     = (bf16*)take((size_t)MTOT * D_MODEL * 2);
  float* attnb  = (float*)take((size_t)MTOT * D_MODEL * 4);
  float* xres   = (float*)take((size_t)MTOT * D_MODEL * 4);
  bf16*  xresb  = (bf16*)take((size_t)MTOT * D_MODEL * 2);
  float* xnorm2 = (float*)take((size_t)MTOT * D_MODEL * 4);
  bf16*  H1     = (bf16*)take((size_t)MTOT * DFF * 2);

  dim3 tb(32, 8);
  transpose_f32_to_bf16<<<dim3(D_MODEL/32, D_MODEL/32), tb, 0, stream>>>(Wq, WqT, D_MODEL, D_MODEL);
  transpose_f32_to_bf16<<<dim3(D_MODEL/32, D_MODEL/32), tb, 0, stream>>>(Wk, WkT, D_MODEL, D_MODEL);
  transpose_f32_to_bf16<<<dim3(D_MODEL/32, D_MODEL/32), tb, 0, stream>>>(Wv, WvT, D_MODEL, D_MODEL);
  transpose_f32_to_bf16<<<dim3(DFF/32,     D_MODEL/32), tb, 0, stream>>>(W1, W1T, D_MODEL, DFF);
  transpose_f32_to_bf16<<<dim3(D_MODEL/32, DFF/32),     tb, 0, stream>>>(W2, W2T, DFF, D_MODEL);

  ln_to_bf16_kernel<<<MTOT, 256, 0, stream>>>(x, g1, be1, xn);

  // Q, K projections (bf16 out), V projection stored transposed per-head
  gemm_bf16_kernel<0><<<dim3(D_MODEL/128, MTOT/128), 128, 0, stream>>>(
      xn, WqT, MTOT, D_MODEL, D_MODEL, nullptr, nullptr, Qb, nullptr);
  gemm_bf16_kernel<0><<<dim3(D_MODEL/128, MTOT/128), 128, 0, stream>>>(
      xn, WkT, MTOT, D_MODEL, D_MODEL, nullptr, nullptr, Kb, nullptr);
  gemm_bf16_kernel<1><<<dim3(D_MODEL/128, MTOT/128), 128, 0, stream>>>(
      xn, WvT, MTOT, D_MODEL, D_MODEL, nullptr, nullptr, Vt, nullptr);

  // flash attention: B*H*(S/16) = 4096 q-tiles, 4 waves/block
  attention_kernel<<<(2 * NHEAD * (SEQ/16)) / 4, 128, 0, stream>>>(Qb, Kb, Vt, attnb);

  resid_ln2_kernel<<<MTOT, 256, 0, stream>>>(attnb, x, g2, be2, xres, xresb, xnorm2);

  // FFN: H1 = relu(x_res @ W1 + b1)  (note: x_res, matching the reference quirk)
  gemm_bf16_kernel<2><<<dim3(DFF/128, MTOT/128), 128, 0, stream>>>(
      xresb, W1T, MTOT, DFF, D_MODEL, b1, nullptr, H1, nullptr);
  // out = H1 @ W2 + b2 + x_norm2
  gemm_bf16_kernel<3><<<dim3(D_MODEL/128, MTOT/128), 128, 0, stream>>>(
      H1, W2T, MTOT, D_MODEL, DFF, b2, xnorm2, nullptr, (float*)d_out);
}
